// GraphormerEdgeLayer_18640158064835
// MI455X (gfx1250) — compile-verified, matrix operand order checked by reading, not run on hardware
//
#include <hip/hip_runtime.h>
#include <hip/hip_bf16.h>
#include <math.h>

// Problem constants (match reference)
constexpr int kE   = 65536;
constexpr int kN   = 16384;
constexpr int kM   = 262144;
constexpr int kD   = 256;
constexpr int kH   = 8;
constexpr int kHID = 1024;

// encoded -inf for the order-preserving float->uint mapping (host-usable constant)
constexpr unsigned kEncNegInf = 0x007FFFFFu;   // ~0xFF800000

// ---------------- CDNA5 WMMA types ----------------
typedef __attribute__((ext_vector_type(16))) __bf16 v16bf;
typedef __attribute__((ext_vector_type(8)))  float  v8f;
typedef unsigned int v4u __attribute__((ext_vector_type(4)));

union FragBF {           // 16 bf16 = 32 bytes = 8 VGPRs
    v4u   u4[2];
    v16bf v;
};

__device__ __forceinline__ unsigned short f2bf(float f) {
    unsigned u = __float_as_uint(f);
    u += 0x7FFFu + ((u >> 16) & 1u);          // round-to-nearest-even
    return (unsigned short)(u >> 16);
}

// order-preserving float<->uint encoding for atomicMax on floats
__device__ __forceinline__ unsigned enc_ord(float f) {
    unsigned u = __float_as_uint(f);
    return (u & 0x80000000u) ? ~u : (u | 0x80000000u);
}
__device__ __forceinline__ float dec_ord(unsigned u) {
    return __uint_as_float((u & 0x80000000u) ? (u & 0x7FFFFFFFu) : ~u);
}

// ---------------- utility kernels ----------------
__global__ void fill_u32_kernel(unsigned* p, unsigned val, long n) {
    long i = (long)blockIdx.x * blockDim.x + threadIdx.x;
    if (i < n) p[i] = val;
}

__global__ void f32_to_bf16_kernel(const float* __restrict__ in,
                                   unsigned short* __restrict__ out, long n) {
    long i = (long)blockIdx.x * blockDim.x + threadIdx.x;
    if (i < n) out[i] = f2bf(in[i]);
}

// W (K x Nc, row-major f32) -> out (Nc x K, row-major bf16)
__global__ void transpose_to_bf16_kernel(const float* __restrict__ W,
                                         int K, int Nc,
                                         unsigned short* __restrict__ out) {
    long idx = (long)blockIdx.x * blockDim.x + threadIdx.x;
    long total = (long)K * Nc;
    if (idx >= total) return;
    int k = (int)(idx / Nc);
    int n = (int)(idx % Nc);
    out[(size_t)n * K + k] = f2bf(W[idx]);
}

// ---------------- LayerNorm: one wave32 per 256-wide row, bf16 out ----------------
__global__ void ln_bf16_kernel(const float* __restrict__ x,
                               const float* __restrict__ sc,
                               const float* __restrict__ bi,
                               unsigned short* __restrict__ out, int rows) {
    int wave = threadIdx.x >> 5;
    int lane = threadIdx.x & 31;
    int row  = blockIdx.x * 8 + wave;
    if (row >= rows) return;
    const float* xr = x + (size_t)row * kD;
    float vals[8];
    float s = 0.f;
#pragma unroll
    for (int j = 0; j < 8; ++j) { vals[j] = xr[lane + 32 * j]; s += vals[j]; }
#pragma unroll
    for (int off = 16; off > 0; off >>= 1) s += __shfl_xor(s, off, 32);
    float mean = s * (1.0f / kD);
    float vs = 0.f;
#pragma unroll
    for (int j = 0; j < 8; ++j) { float d = vals[j] - mean; vs += d * d; }
#pragma unroll
    for (int off = 16; off > 0; off >>= 1) vs += __shfl_xor(vs, off, 32);
    float inv = rsqrtf(vs * (1.0f / kD) + 1e-5f);
    unsigned short* orow = out + (size_t)row * kD;
#pragma unroll
    for (int j = 0; j < 8; ++j) {
        int c = lane + 32 * j;
        orow[c] = f2bf((vals[j] - mean) * inv * sc[c] + bi[c]);
    }
}

// ---------------- WMMA GEMM ----------------
// C(Mrows x Nc) = act(A16(Mrows x K) @ Bt16(Nc x K)^T + bias) + resid
// block: 256 threads = 8 waves, computes 128 rows x 64 cols
// wave: 32x32 tile via 4 accumulators, 4 x v_wmma_f32_16x16x32_bf16 per K-step
__device__ __forceinline__ FragBF load_frag(const unsigned short* p) {
    FragBF f;
    f.u4[0] = *(const v4u*)(p);
    f.u4[1] = *(const v4u*)(p + 16);
    return f;
}

__global__ __launch_bounds__(256)
void gemm_bf16_wmma(const unsigned short* __restrict__ A16,
                    const unsigned short* __restrict__ Bt16,
                    const float* __restrict__ bias,
                    const float* __restrict__ resid,
                    float* __restrict__ outf,
                    unsigned short* __restrict__ out16,
                    int K, int Nc, int act) {
    const int wave  = threadIdx.x >> 5;
    const int lane  = threadIdx.x & 31;
    const int wm    = wave >> 1;
    const int wn    = wave & 1;
    const int rbase = blockIdx.x * 128 + wm * 32;
    const int cbase = blockIdx.y * 64 + wn * 32;
    const int half  = lane >> 4;     // 0: K=+0..7/+16..23, 1: K=+8..15/+24..31
    const int l15   = lane & 15;
    const int kb    = half * 8;

    v8f c00 = {}, c01 = {}, c10 = {}, c11 = {};

    const unsigned short* pa0 = A16 + (size_t)(rbase + l15) * K + kb;
    const unsigned short* pa1 = pa0 + (size_t)16 * K;
    const unsigned short* pb0 = Bt16 + (size_t)(cbase + l15) * K + kb;
    const unsigned short* pb1 = pb0 + (size_t)16 * K;

    for (int k0 = 0; k0 < K; k0 += 32) {
        FragBF a0 = load_frag(pa0 + k0);
        FragBF a1 = load_frag(pa1 + k0);
        FragBF b0 = load_frag(pb0 + k0);
        FragBF b1 = load_frag(pb1 + k0);
        __builtin_prefetch(pa0 + k0 + 32, 0, 1);   // global_prefetch_b8
        __builtin_prefetch(pb0 + k0 + 32, 0, 1);
        c00 = __builtin_amdgcn_wmma_f32_16x16x32_bf16(false, a0.v, false, b0.v, (short)0, c00, false, false);
        c01 = __builtin_amdgcn_wmma_f32_16x16x32_bf16(false, a0.v, false, b1.v, (short)0, c01, false, false);
        c10 = __builtin_amdgcn_wmma_f32_16x16x32_bf16(false, a1.v, false, b0.v, (short)0, c10, false, false);
        c11 = __builtin_amdgcn_wmma_f32_16x16x32_bf16(false, a1.v, false, b1.v, (short)0, c11, false, false);
    }

#pragma unroll
    for (int mi = 0; mi < 2; ++mi) {
#pragma unroll
        for (int ni = 0; ni < 2; ++ni) {
            v8f acc = (mi == 0) ? (ni == 0 ? c00 : c01) : (ni == 0 ? c10 : c11);
            int col = cbase + ni * 16 + l15;
            float bb = bias ? bias[col] : 0.0f;
#pragma unroll
            for (int r = 0; r < 8; ++r) {
                int row = rbase + mi * 16 + half * 8 + r;
                float v = acc[r] + bb;
                if (act) v = 0.5f * v * (1.0f + erff(v * 0.70710678118654752f)); // exact GELU
                size_t o = (size_t)row * Nc + col;
                if (resid) v += resid[o];
                if (outf)  outf[o]  = v;
                if (out16) out16[o] = f2bf(v);
            }
        }
    }
}

// ---------------- sparse attention passes ----------------
__global__ void attn_logits_kernel(const float* __restrict__ Q,
                                   const float* __restrict__ Kmat,
                                   const int* __restrict__ src_e,
                                   const int* __restrict__ dst_e,
                                   const float* __restrict__ eebias,
                                   float* __restrict__ LOG,
                                   unsigned* __restrict__ SM,
                                   float* __restrict__ DEG) {
    int idx = blockIdx.x * blockDim.x + threadIdx.x;
    if (idx >= kM * kH) return;
    int m = idx >> 3, h = idx & 7;
    int dn = dst_e[m], sn = src_e[m];
    const float4* qp = (const float4*)(Q    + (size_t)dn * kD + h * 32);
    const float4* kp = (const float4*)(Kmat + (size_t)sn * kD + h * 32);
    float dot = 0.f;
#pragma unroll
    for (int j = 0; j < 8; ++j) {
        float4 a = qp[j], b = kp[j];
        dot += a.x * b.x + a.y * b.y + a.z * b.z + a.w * b.w;
    }
    float logit = dot * 0.17677669529663687f + eebias[idx];   // HD^-0.5
    LOG[idx] = logit;
    atomicMax(SM + (size_t)dn * kH + h, enc_ord(logit));
    if (h == 0) atomicAdd(DEG + dn, 1.0f);
}

__global__ void attn_exp_kernel(float* __restrict__ LOG,
                                const unsigned* __restrict__ SM,
                                float* __restrict__ SS,
                                const int* __restrict__ dst_e) {
    int idx = blockIdx.x * blockDim.x + threadIdx.x;
    if (idx >= kM * kH) return;
    int m = idx >> 3, h = idx & 7;
    int dn = dst_e[m];
    float ex = expf(LOG[idx] - dec_ord(SM[(size_t)dn * kH + h]));
    LOG[idx] = ex;
    atomicAdd(SS + (size_t)dn * kH + h, ex);
}

__global__ void attn_scatter_kernel(const float* __restrict__ LOG,
                                    const float* __restrict__ SS,
                                    const float* __restrict__ V,
                                    const int* __restrict__ src_e,
                                    const int* __restrict__ dst_e,
                                    float* __restrict__ agg) {
    long idx = (long)blockIdx.x * blockDim.x + threadIdx.x;   // exactly kM*kD threads
    int m = (int)(idx >> 8);
    int d = (int)(idx & 255);
    int h = d >> 5;
    int dn = dst_e[m], sn = src_e[m];
    float attn = LOG[(size_t)m * kH + h] / (SS[(size_t)dn * kH + h] + 1e-16f);
    atomicAdd(agg + (size_t)dn * kD + d, V[(size_t)sn * kD + d] * attn);
}

// ---------------- degree scaling + node scatter ----------------
__global__ void deg_scale_kernel(const float* __restrict__ pre,
                                 const float* __restrict__ DEG,
                                 const float* __restrict__ coef,
                                 float* __restrict__ outE) {
    long idx = (long)blockIdx.x * blockDim.x + threadIdx.x;   // exactly kE*kD
    int e = (int)(idx >> 8);
    int d = (int)(idx & 255);
    float p  = pre[idx];
    float ld = log1pf(DEG[e]);
    outE[idx] = p * coef[2 * d] + p * ld * coef[2 * d + 1];
}

__global__ void node_scatter_kernel(const float* __restrict__ eo,
                                    const int* __restrict__ eidx,
                                    float* __restrict__ nmsg) {
    long idx = (long)blockIdx.x * blockDim.x + threadIdx.x;   // exactly kE*kD
    int e = (int)(idx >> 8);
    int d = (int)(idx & 255);
    float v = eo[idx];
    atomicAdd(nmsg + (size_t)eidx[e]      * kD + d, v);
    atomicAdd(nmsg + (size_t)eidx[kE + e] * kD + d, v);
}

// ---------------- launch ----------------
extern "C" void kernel_launch(void* const* d_in, const int* in_sizes, int n_in,
                              void* d_out, int out_size, void* d_ws, size_t ws_size,
                              hipStream_t stream) {
    (void)in_sizes; (void)n_in; (void)out_size; (void)ws_size;

    const float* edge_attr  = (const float*)d_in[0];
    const float* x          = (const float*)d_in[1];
    const int*   src_e      = (const int*)d_in[2];
    const int*   dst_e      = (const int*)d_in[3];
    const int*   edge_index = (const int*)d_in[4];
    const float* ee_bias    = (const float*)d_in[5];
    const float* ln1_s = (const float*)d_in[6];
    const float* ln1_b = (const float*)d_in[7];
    const float* Wq = (const float*)d_in[8];   const float* bq = (const float*)d_in[9];
    const float* Wk = (const float*)d_in[10];  const float* bk = (const float*)d_in[11];
    const float* Wv = (const float*)d_in[12];  const float* bv = (const float*)d_in[13];
    const float* Wo = (const float*)d_in[14];  const float* bo = (const float*)d_in[15];
    const float* ln2_s = (const float*)d_in[16];
    const float* ln2_b = (const float*)d_in[17];
    const float* W1 = (const float*)d_in[18];  const float* b1 = (const float*)d_in[19];
    const float* W2 = (const float*)d_in[20];  const float* b2 = (const float*)d_in[21];
    const float* deg_coef = (const float*)d_in[22];
    const float* We2n = (const float*)d_in[23]; const float* be2n = (const float*)d_in[24];

    // ---- workspace arena (5 big slots of E*D*4 bytes, reused across phases) ----
    char* ws = (char*)d_ws;
    const size_t SLOT = (size_t)kE * kD * 4;   // 64 MiB
    float*          S1f = (float*)(ws);
    unsigned short* S1h = (unsigned short*)(ws);
    float*          S2f = (float*)(ws + SLOT);
    unsigned short* S2h = (unsigned short*)(ws + SLOT);
    float*          S3f = (float*)(ws + 2 * SLOT);
    unsigned short* S3h = (unsigned short*)(ws + 2 * SLOT);   // T16 spans S3+S4
    float*          S4f = (float*)(ws + 3 * SLOT);
    float*          S5f = (float*)(ws + 4 * SLOT);
    unsigned short* S5h = (unsigned short*)(ws + 4 * SLOT);
    float*    LOG  = (float*)(ws + 5 * SLOT);                       // M*H f32
    unsigned* SM   = (unsigned*)((char*)LOG + (size_t)kM * kH * 4); // E*H
    float*    SS   = (float*)((char*)SM + (size_t)kE * kH * 4);     // E*H
    float*    DEG  = (float*)((char*)SS + (size_t)kE * kH * 4);     // E
    float*    NMSG = (float*)((char*)DEG + (size_t)kE * 4);         // N*D
    unsigned short* WTS = (unsigned short*)((char*)NMSG + (size_t)kN * kD * 4);
    unsigned short* Wq_t   = WTS;
    unsigned short* Wk_t   = Wq_t + (size_t)kD * kD;
    unsigned short* Wv_t   = Wk_t + (size_t)kD * kD;
    unsigned short* Wo_t   = Wv_t + (size_t)kD * kD;
    unsigned short* W1_t   = Wo_t + (size_t)kD * kD;      // HID x D
    unsigned short* W2_t   = W1_t + (size_t)kD * kHID;    // D x HID
    unsigned short* We2n_t = W2_t + (size_t)kHID * kD;

    float* outE    = (float*)d_out;
    float* outNode = outE + (size_t)kE * kD;

    const int T = 256;
    auto blocks = [](long n) { return dim3((unsigned)((n + 255) / 256)); };

    // 0) weights -> bf16, transposed to N x K
    transpose_to_bf16_kernel<<<blocks((long)kD * kD), T, 0, stream>>>(Wq, kD, kD, Wq_t);
    transpose_to_bf16_kernel<<<blocks((long)kD * kD), T, 0, stream>>>(Wk, kD, kD, Wk_t);
    transpose_to_bf16_kernel<<<blocks((long)kD * kD), T, 0, stream>>>(Wv, kD, kD, Wv_t);
    transpose_to_bf16_kernel<<<blocks((long)kD * kD), T, 0, stream>>>(Wo, kD, kD, Wo_t);
    transpose_to_bf16_kernel<<<blocks((long)kD * kHID), T, 0, stream>>>(W1, kD, kHID, W1_t);
    transpose_to_bf16_kernel<<<blocks((long)kHID * kD), T, 0, stream>>>(W2, kHID, kD, W2_t);
    transpose_to_bf16_kernel<<<blocks((long)kD * kD), T, 0, stream>>>(We2n, kD, kD, We2n_t);

    // init accumulators
    fill_u32_kernel<<<blocks((long)kE * kH), T, 0, stream>>>(SM, kEncNegInf, (long)kE * kH); // enc(-inf)
    fill_u32_kernel<<<blocks((long)kE * kH), T, 0, stream>>>((unsigned*)SS, 0u, (long)kE * kH);
    fill_u32_kernel<<<blocks((long)kE), T, 0, stream>>>((unsigned*)DEG, 0u, (long)kE);
    fill_u32_kernel<<<blocks((long)kE * kD), T, 0, stream>>>((unsigned*)S5f, 0u, (long)kE * kD);     // agg
    fill_u32_kernel<<<blocks((long)kN * kD), T, 0, stream>>>((unsigned*)NMSG, 0u, (long)kN * kD);

    // 1) LN1 -> en16 (S1)
    ln_bf16_kernel<<<kE / 8, T, 0, stream>>>(edge_attr, ln1_s, ln1_b, S1h, kE);

    // 2) QKV GEMMs (en16 @ W*) -> f32 Q(S2), K(S3), V(S4)
    dim3 gEdge(kE / 128, kD / 64);
    gemm_bf16_wmma<<<gEdge, T, 0, stream>>>(S1h, Wq_t, bq, nullptr, S2f, nullptr, kD, kD, 0);
    gemm_bf16_wmma<<<gEdge, T, 0, stream>>>(S1h, Wk_t, bk, nullptr, S3f, nullptr, kD, kD, 0);
    gemm_bf16_wmma<<<gEdge, T, 0, stream>>>(S1h, Wv_t, bv, nullptr, S4f, nullptr, kD, kD, 0);

    // 3) sparse segment softmax attention
    attn_logits_kernel<<<blocks((long)kM * kH), T, 0, stream>>>(S2f, S3f, src_e, dst_e, ee_bias, LOG, SM, DEG);
    attn_exp_kernel<<<blocks((long)kM * kH), T, 0, stream>>>(LOG, SM, SS, dst_e);
    attn_scatter_kernel<<<blocks((long)kM * kD), T, 0, stream>>>(LOG, SS, S4f, src_e, dst_e, S5f);

    // 4) agg -> bf16 (into S1, en16 is dead)
    f32_to_bf16_kernel<<<blocks((long)kE * kD), T, 0, stream>>>(S5f, S1h, (long)kE * kD);

    // 5) h1 = edge_attr + agg16 @ Wo + bo  -> S2 (f32)
    gemm_bf16_wmma<<<gEdge, T, 0, stream>>>(S1h, Wo_t, bo, edge_attr, S2f, nullptr, kD, kD, 0);

    // 6) LN2(h1) -> h1n16 (S5, agg f32 dead)
    ln_bf16_kernel<<<kE / 8, T, 0, stream>>>(S2f, ln2_s, ln2_b, S5h, kE);

    // 7) T16 = gelu(h1n16 @ W1 + b1) -> bf16 in S3+S4 (E x HID)
    dim3 gFF1(kE / 128, kHID / 64);
    gemm_bf16_wmma<<<gFF1, T, 0, stream>>>(S5h, W1_t, b1, nullptr, nullptr, S3h, kD, kHID, 1);

    // 8) pre = h1 + T16 @ W2 + b2 -> S1 (f32)
    gemm_bf16_wmma<<<gEdge, T, 0, stream>>>(S3h, W2_t, b2, S2f, S1f, nullptr, kHID, kD, 0);

    // 9) degree scaling -> final edge_out in d_out
    deg_scale_kernel<<<blocks((long)kE * kD), T, 0, stream>>>(S1f, DEG, deg_coef, outE);

    // 10) node message scatter (reads final edge_out)
    node_scatter_kernel<<<blocks((long)kE * kD), T, 0, stream>>>(outE, edge_index, NMSG);

    // 11) node_msg -> bf16 (S2, h1 dead)
    f32_to_bf16_kernel<<<blocks((long)kN * kD), T, 0, stream>>>(NMSG, S2h, (long)kN * kD);

    // 12) node_out = x + nm16 @ We2n + be2n
    dim3 gNode(kN / 128, kD / 64);
    gemm_bf16_wmma<<<gNode, T, 0, stream>>>(S2h, We2n_t, be2n, x, outNode, nullptr, kD, kD, 0);
}
